// AdaptiveSoftmax_58729382805612
// MI455X (gfx1250) — matrix operand order repaired and checked by reading, not compiled, not run on hardware
//
#include <hip/hip_runtime.h>
#include <hip/hip_bf16.h>

// ---------------- types for WMMA ----------------
typedef __attribute__((ext_vector_type(16))) __bf16 v16bf;
typedef __attribute__((ext_vector_type(8)))  __bf16 v8bf;
typedef __attribute__((ext_vector_type(8)))  float  v8f;

#define VOCAB   50257
#define HEADSZ  10002
#define ROWS    2048      // B*T
#define DIM     1024
#define T0N     40000     // CUTOFF[1]-CUTOFF[0]
#define T1N     257       // CUTOFF[2]-CUTOFF[1]
#define HEAD_NPAD 10048   // 10002 -> multiple of 64
#define T1_NPAD   320     // 257   -> multiple of 64

// ---------------- f32 -> bf16 conversion (with zero padding) ----------------
__global__ __launch_bounds__(256)
void cvt_bf16_kernel(const float* __restrict__ src, __bf16* __restrict__ dst,
                     long nsrc, long ntot) {
    long i = (long)blockIdx.x * blockDim.x + threadIdx.x;
    long stride = (long)gridDim.x * blockDim.x;
    for (; i < ntot; i += stride)
        dst[i] = (i < nsrc) ? (__bf16)src[i] : (__bf16)0.0f;
}

// ---------------- bf16 GEMM:  C[M x N] = A[M x K] * W[N x K]^T ----------------
// One wave: 32(M) x 64(N) -> 8 accumulators, each B fragment reused for 2 A
// fragments (doubled arithmetic intensity vs 16x64). Block: 8 waves stacked in
// M -> 256M x 64N. Requires: M % 256 == 0, Npad % 64 == 0, K % 32 == 0.
union AFrag { v16bf v; struct { v8bf lo, hi; } p; };

template <typename OutT>
__global__ __launch_bounds__(256)
void gemm_bf16_nt(const __bf16* __restrict__ A, const __bf16* __restrict__ W,
                  OutT* __restrict__ C, int K, int Nstore, long ldc) {
    const int lane = threadIdx.x & 31;
    const int wave = threadIdx.x >> 5;
    const int m0   = blockIdx.y * 256 + wave * 32;
    const int n0   = blockIdx.x * 64;
    const int l16  = lane & 15;
    const int hi   = lane >> 4;      // 0 | 1
    const int koA  = hi * 16;        // A: elems 0..7 = K[koA..], elems 8..15 = K[koA+8..]
    const int koB  = hi * 16;        // B: elems 0..15 = K[koB .. koB+15]

    const __bf16* aRow0 = A + (size_t)(m0 + l16) * K;
    const __bf16* aRow1 = aRow0 + (size_t)16 * K;
    const __bf16* bRow0 = W + (size_t)(n0 + l16) * K;

    v8f acc0[4] = {};
    v8f acc1[4] = {};

    for (int kk = 0; kk < K; kk += 32) {
        // prefetch next K-block of the streaming B operand into cache
        if (kk + 32 < K)
            __builtin_prefetch(bRow0 + (kk + 32) + koB, 0, 1);

        AFrag a0, a1;
        a0.p.lo = *(const v8bf*)(aRow0 + kk + koA);
        a0.p.hi = *(const v8bf*)(aRow0 + kk + koA + 8);
        a1.p.lo = *(const v8bf*)(aRow1 + kk + koA);
        a1.p.hi = *(const v8bf*)(aRow1 + kk + koA + 8);

        #pragma unroll
        for (int i = 0; i < 4; ++i) {
            v16bf b = *(const v16bf*)(bRow0 + (size_t)(16 * i) * K + kk + koB);
            acc0[i] = __builtin_amdgcn_wmma_f32_16x16x32_bf16(
                false, a0.v, false, b, (short)0, acc0[i], false, false);
            acc1[i] = __builtin_amdgcn_wmma_f32_16x16x32_bf16(
                false, a1.v, false, b, (short)0, acc1[i], false, false);
        }
    }

    // C/D layout: VGPR r holds M = base + r + hi*8, column = n0 + 16*i + l16
    #pragma unroll
    for (int i = 0; i < 4; ++i) {
        int c = n0 + 16 * i + l16;
        if (c < Nstore) {
            #pragma unroll
            for (int r = 0; r < 8; ++r) {
                int row = m0 + r + hi * 8;
                C[(size_t)row * ldc + c] = (OutT)acc0[i][r];
                C[(size_t)(row + 16) * ldc + c] = (OutT)acc1[i][r];
            }
        }
    }
}

// ---------------- online logsumexp block reduction helper ----------------
__device__ inline float block_lse_finish(float m, float s) {
    __shared__ float sm[256], ss[256];
    int tid = threadIdx.x;
    sm[tid] = m; ss[tid] = s;
    __syncthreads();
    for (int off = 128; off > 0; off >>= 1) {
        if (tid < off) {
            float m2 = sm[tid + off], s2 = ss[tid + off];
            float M  = fmaxf(sm[tid], m2);
            ss[tid]  = ss[tid] * __expf(sm[tid] - M) + s2 * __expf(m2 - M);
            sm[tid]  = M;
        }
        __syncthreads();
    }
    return sm[0] + __logf(ss[0]);
}

__device__ inline void lse_accum(float x, float& m, float& s) {
    if (x > m) { s = s * __expf(m - x) + 1.0f; m = x; }
    else       { s += __expf(x - m); }
}

// ---------------- head finalize: log_softmax in place + save priors/overlap ----------------
__global__ __launch_bounds__(256)
void head_finalize_kernel(float* __restrict__ out, float* __restrict__ sv) {
    const int row = blockIdx.x;
    float* rowp = out + (size_t)row * VOCAB;
    const int tid = threadIdx.x;

    float m = -INFINITY, s = 0.0f;
    for (int c = tid; c < HEADSZ; c += 256) lse_accum(rowp[c], m, s);
    float lse = block_lse_finish(m, s);

    // sv: [0]=lp[9999] (tail0 prior), [1]=lp[10000] (tail1 prior / restore), [2]=lp[10001] (restore)
    if (tid < 3) sv[row * 3 + tid] = rowp[9999 + tid] - lse;
    __syncthreads();

    for (int c = tid; c < HEADSZ; c += 256) rowp[c] -= lse;
}

// ---------------- tail0 finalize: columns [10000, 50000) ----------------
__global__ __launch_bounds__(256)
void tail0_finalize_kernel(float* __restrict__ out, const float* __restrict__ sv,
                           const int* __restrict__ tgt) {
    const int row = blockIdx.x;
    float* rowp = out + (size_t)row * VOCAB + 10000;
    const int tid = threadIdx.x;
    const int t = tgt[row];
    const bool mask = (t >= 10000) && (t < 50000);

    if (!mask) {
        // restore head log-probs in the 2 overlapped columns, zero elsewhere
        for (int c = tid; c < T0N; c += 256) {
            float v = 0.0f;
            if (c == 0)      v = sv[row * 3 + 1];
            else if (c == 1) v = sv[row * 3 + 2];
            rowp[c] = v;
        }
        return;
    }

    float m = -INFINITY, s = 0.0f;
    for (int c = tid; c < T0N; c += 256) lse_accum(rowp[c], m, s);
    float lse = block_lse_finish(m, s);
    float prior = sv[row * 3 + 0];
    for (int c = tid; c < T0N; c += 256) rowp[c] = rowp[c] - lse + prior;
}

// ---------------- tail1 finalize: columns [50000, 50257) ----------------
__global__ __launch_bounds__(256)
void tail1_finalize_kernel(float* __restrict__ out, const float* __restrict__ logits,
                           const float* __restrict__ sv, const int* __restrict__ tgt) {
    const int row = blockIdx.x;
    float* rowp = out + (size_t)row * VOCAB + 50000;
    const float* lg = logits + (size_t)row * T1_NPAD;
    const int tid = threadIdx.x;
    const int t = tgt[row];
    const bool mask = (t >= 50000) && (t < VOCAB);

    if (!mask) {
        for (int c = tid; c < T1N; c += 256) rowp[c] = 0.0f;
        return;
    }

    float m = -INFINITY, s = 0.0f;
    for (int c = tid; c < T1N; c += 256) lse_accum(lg[c], m, s);
    float lse = block_lse_finish(m, s);
    float prior = sv[row * 3 + 1];
    for (int c = tid; c < T1N; c += 256) rowp[c] = lg[c] - lse + prior;
}

// ---------------- host side ----------------
static inline size_t align256(size_t x) { return (x + 255) & ~(size_t)255; }

extern "C" void kernel_launch(void* const* d_in, const int* in_sizes, int n_in,
                              void* d_out, int out_size, void* d_ws, size_t ws_size,
                              hipStream_t stream) {
    const float* xi     = (const float*)d_in[0];   // (2,1024,1024) f32
    const int*   tgt    = (const int*)  d_in[1];   // (2,1024)      i32
    const float* head_w = (const float*)d_in[2];   // (10002,1024)
    const float* t0_a   = (const float*)d_in[3];   // (1024,1024)
    const float* t0_b   = (const float*)d_in[4];   // (40000,1024)
    const float* t1_a   = (const float*)d_in[5];   // (256,1024)
    const float* t1_b   = (const float*)d_in[6];   // (257,256)
    float* out = (float*)d_out;

    // workspace carve-up (bf16 copies + small scratch), ~118 MB
    char* p = (char*)d_ws;
    size_t off = 0;
    __bf16* xiB    = (__bf16*)(p + off); off = align256(off + (size_t)ROWS * DIM * 2);
    __bf16* headwB = (__bf16*)(p + off); off = align256(off + (size_t)HEAD_NPAD * DIM * 2);
    __bf16* t0aB   = (__bf16*)(p + off); off = align256(off + (size_t)DIM * DIM * 2);
    __bf16* t0bB   = (__bf16*)(p + off); off = align256(off + (size_t)T0N * DIM * 2);
    __bf16* t1aB   = (__bf16*)(p + off); off = align256(off + (size_t)256 * DIM * 2);
    __bf16* t1bB   = (__bf16*)(p + off); off = align256(off + (size_t)T1_NPAD * 256 * 2);
    __bf16* h0B    = (__bf16*)(p + off); off = align256(off + (size_t)ROWS * DIM * 2);
    __bf16* h1B    = (__bf16*)(p + off); off = align256(off + (size_t)ROWS * 256 * 2);
    float*  t1log  = (float*) (p + off); off = align256(off + (size_t)ROWS * T1_NPAD * 4);
    float*  sv     = (float*) (p + off); off = align256(off + (size_t)ROWS * 3 * 4);

    // 1) convert everything to bf16 (zero-pad N to multiples of 64)
    dim3 cb(256), cg(2048);
    cvt_bf16_kernel<<<cg, cb, 0, stream>>>(xi,     xiB,    (long)ROWS * DIM,   (long)ROWS * DIM);
    cvt_bf16_kernel<<<cg, cb, 0, stream>>>(head_w, headwB, (long)HEADSZ * DIM, (long)HEAD_NPAD * DIM);
    cvt_bf16_kernel<<<cg, cb, 0, stream>>>(t0_a,   t0aB,   (long)DIM * DIM,    (long)DIM * DIM);
    cvt_bf16_kernel<<<cg, cb, 0, stream>>>(t0_b,   t0bB,   (long)T0N * DIM,    (long)T0N * DIM);
    cvt_bf16_kernel<<<cg, cb, 0, stream>>>(t1_a,   t1aB,   (long)256 * DIM,    (long)256 * DIM);
    cvt_bf16_kernel<<<cg, cb, 0, stream>>>(t1_b,   t1bB,   (long)T1N * 256,    (long)T1_NPAD * 256);

    dim3 gb(256);

    // 2) head logits -> out[:, :10002]
    gemm_bf16_nt<float><<<dim3(HEAD_NPAD / 64, ROWS / 256), gb, 0, stream>>>(
        xiB, headwB, out, DIM, HEADSZ, (long)VOCAB);

    // 3) head log_softmax in place, save priors / overlap columns
    head_finalize_kernel<<<ROWS, 256, 0, stream>>>(out, sv);

    // 4) tail0: h0 = xi @ t0_a^T (bf16), logits = h0 @ t0_b^T -> out[:, 10000:50000]
    gemm_bf16_nt<__bf16><<<dim3(DIM / 64, ROWS / 256), gb, 0, stream>>>(
        xiB, t0aB, h0B, DIM, DIM, (long)DIM);
    gemm_bf16_nt<float><<<dim3(T0N / 64, ROWS / 256), gb, 0, stream>>>(
        h0B, t0bB, out + 10000, DIM, T0N, (long)VOCAB);
    tail0_finalize_kernel<<<ROWS, 256, 0, stream>>>(out, sv, tgt);

    // 5) tail1: h1 = xi @ t1_a^T (bf16), logits = h1 @ t1_b^T -> ws, finalize -> out[:, 50000:]
    gemm_bf16_nt<__bf16><<<dim3(256 / 64, ROWS / 256), gb, 0, stream>>>(
        xiB, t1aB, h1B, DIM, 256, (long)256);
    gemm_bf16_nt<float><<<dim3(T1_NPAD / 64, ROWS / 256), gb, 0, stream>>>(
        h1B, t1bB, t1log, 256, T1_NPAD, (long)T1_NPAD);
    tail1_finalize_kernel<<<ROWS, 256, 0, stream>>>(out, t1log, sv, tgt);

    (void)in_sizes; (void)n_in; (void)out_size; (void)ws_size;
}
